// QMoEEncoder_84172769067212
// MI455X (gfx1250) — compile-verified, matrix operand order checked by reading, not run on hardware
//
#include <hip/hip_runtime.h>
#include <hip/hip_bf16.h>

typedef unsigned short u16;
typedef unsigned int   u32;
typedef unsigned long long u64;

typedef __attribute__((ext_vector_type(16))) __bf16 v16bf;
typedef __attribute__((ext_vector_type(8)))  float  v8f;

#define Bsz 8192
#define Dd  1024
#define Hh  1024
#define Ee  8
#define NQq 8
#define NLl 2
#define NBb 2
#define KCAT (Ee * Hh)   // 8192

__device__ __forceinline__ u16 f2bf(float f) {
    u32 u = __float_as_uint(f);
    u32 r = u + 0x7FFFu + ((u >> 16) & 1u);
    return (u16)(r >> 16);
}

// ---------------------------------------------------------------------------
// Kernel 0: w2 (NB,E,H,D) fp32 -> bf16, N-major [blk][d][e*H+h], via LDS
// 64x64 tile transpose so both global streams are coalesced.
// ---------------------------------------------------------------------------
__global__ __launch_bounds__(256) void conv_w2_kernel(
    const float* __restrict__ w2, u16* __restrict__ w2bf)
{
    __shared__ u16 t[64][72];            // padded vs bank conflicts
    const int tid   = threadIdx.x;
    const int be    = blockIdx.z;        // blk*8 + e
    const int hBase = blockIdx.y * 64;
    const int dBase = blockIdx.x * 64;
    const int blk = be >> 3, e = be & 7;

    const float* src = w2 + (size_t)be * Hh * Dd;
    for (int i = tid; i < 64 * 64; i += 256) {
        int h = i >> 6, d = i & 63;      // consecutive tid -> consecutive d
        t[d][h] = f2bf(src[(size_t)(hBase + h) * Dd + dBase + d]);
    }
    __syncthreads();
    u16* dst = w2bf + (size_t)blk * Dd * KCAT;
    for (int i = tid; i < 64 * 64; i += 256) {
        int d = i >> 6, h = i & 63;      // consecutive tid -> consecutive h
        dst[(size_t)(dBase + d) * KCAT + e * Hh + hBase + h] = t[d][h];
    }
}

// ---------------------------------------------------------------------------
// Kernel 1: per-token gate + top-2 + 8-qubit statevector sim + H_cat row.
// One workgroup (256 thr = 8 wave32) per token. One amplitude per thread.
// ---------------------------------------------------------------------------
__global__ __launch_bounds__(256) void gate_qsim_h_kernel(
    const float* __restrict__ xin,   // [B][D]
    const float* __restrict__ gW,    // [D][E]
    const float* __restrict__ gb,    // [E]
    const float* __restrict__ qp,    // [E][NL][NQ][3]
    const float* __restrict__ rW,    // [E][NQ][H]
    const float* __restrict__ rb,    // [E][H]
    u16*  __restrict__ Hcat,         // [B][E*H] bf16
    float* __restrict__ svals)       // [B][E]
{
    const int b    = blockIdx.x;
    const int tid  = threadIdx.x;
    const int lane = tid & 31;
    const int wave = tid >> 5;

    __shared__ float  xs[Dd];
    __shared__ float2 st[2][256];
    __shared__ float  logit[Ee];
    __shared__ float  sv[Ee];
    __shared__ float  qv[2][NQq];
    __shared__ int    sel[2];

    for (int d = tid; d < Dd; d += 256) xs[d] = xin[(size_t)b * Dd + d];
    __syncthreads();

    // ---- gate logits: wave e computes logit[e]
    {
        const int e = wave;
        float acc = 0.f;
        for (int d = lane; d < Dd; d += 32) acc += xs[d] * gW[d * Ee + e];
        #pragma unroll
        for (int off = 16; off; off >>= 1) acc += __shfl_xor(acc, off, 32);
        if (lane == 0) logit[e] = acc + gb[e];
    }
    __syncthreads();

    // ---- softmax / top-2 / re-softmax (tiny; one thread)
    if (tid == 0) {
        float mx = logit[0];
        for (int e = 1; e < Ee; ++e) mx = fmaxf(mx, logit[e]);
        float ex[Ee], ssum = 0.f;
        for (int e = 0; e < Ee; ++e) { ex[e] = __expf(logit[e] - mx); ssum += ex[e]; }
        float p[Ee];
        for (int e = 0; e < Ee; ++e) p[e] = ex[e] / ssum;
        int i1 = 0;
        for (int e = 1; e < Ee; ++e) if (p[e] > p[i1]) i1 = e;       // first index on ties
        int i2 = (i1 == 0) ? 1 : 0;
        for (int e = 0; e < Ee; ++e) if (e != i1 && p[e] > p[i2]) i2 = e;
        float m2 = fmaxf(p[i1], p[i2]);
        float e1 = __expf(p[i1] - m2), e2 = __expf(p[i2] - m2);
        float z = e1 + e2;
        for (int e = 0; e < Ee; ++e) sv[e] = 0.f;
        sv[i1] = e1 / z; sv[i2] = e2 / z;
        sel[0] = i1; sel[1] = i2;
    }
    __syncthreads();

    // ---- statevector sim for the two selected experts
    const int idx = tid;   // amplitude index 0..255; wire q -> bit (7-q)
    for (int slot = 0; slot < 2; ++slot) {
        const int e = sel[slot];
        const float* qpe = qp + (size_t)e * NLl * NQq * 3;

        float amp = 1.f;
        #pragma unroll
        for (int q = 0; q < NQq; ++q) {
            float a = xs[q] * 0.5f;
            amp *= ((idx >> (7 - q)) & 1) ? __sinf(a) : __cosf(a);
        }
        int cur = 0;
        st[cur][idx] = make_float2(amp, 0.f);
        __syncthreads();

        for (int l = 0; l < NLl; ++l) {
            for (int q = 0; q < NQq; ++q) {
                const float phi = qpe[(l * NQq + q) * 3 + 0];
                const float th  = qpe[(l * NQq + q) * 3 + 1];
                const float om  = qpe[(l * NQq + q) * 3 + 2];
                const float c2 = __cosf(th * 0.5f), s2 = __sinf(th * 0.5f);
                const float ap = 0.5f * (phi + om), am = 0.5f * (phi - om);
                const float cap = __cosf(ap), sap = __sinf(ap);
                const float cam = __cosf(am), sam = __sinf(am);
                // U00 = e^{-i ap} c2 ; U01 = -e^{+i am} s2
                // U10 = e^{-i am} s2 ; U11 = e^{+i ap} c2
                const float U00r =  cap * c2, U00i = -sap * c2;
                const float U01r = -cam * s2, U01i = -sam * s2;
                const float U10r =  cam * s2, U10i = -sam * s2;
                const float U11r =  cap * c2, U11i =  sap * c2;

                const int bitpos = 7 - q;
                const int bq = (idx >> bitpos) & 1;
                const int partner = idx ^ (1 << bitpos);
                float2 a0 = st[cur][bq ? partner : idx];
                float2 a1 = st[cur][bq ? idx : partner];
                float2 r;
                if (!bq) {
                    r.x = U00r * a0.x - U00i * a0.y + U01r * a1.x - U01i * a1.y;
                    r.y = U00r * a0.y + U00i * a0.x + U01r * a1.y + U01i * a1.x;
                } else {
                    r.x = U10r * a0.x - U10i * a0.y + U11r * a1.x - U11i * a1.y;
                    r.y = U10r * a0.y + U10i * a0.x + U11r * a1.y + U11i * a1.x;
                }
                st[1 - cur][idx] = r;
                __syncthreads();
                cur ^= 1;
            }
            for (int q = 0; q < NQq - 1; ++q) {      // CNOT chain (q, q+1)
                const int cpos = 7 - q, tpos = 7 - (q + 1);
                const int cbit = (idx >> cpos) & 1;
                const int src  = cbit ? (idx ^ (1 << tpos)) : idx;
                float2 r = st[cur][src];
                st[1 - cur][idx] = r;
                __syncthreads();
                cur ^= 1;
            }
        }

        // probs + Z expectations (wave k reduces observable k)
        float2 v = st[cur][idx];
        st[1 - cur][idx].x = v.x * v.x + v.y * v.y;
        __syncthreads();
        {
            const int k = wave;
            float acc = 0.f;
            for (int i = lane; i < 256; i += 32) {
                float sgn = ((i >> (7 - k)) & 1) ? -1.f : 1.f;
                acc += st[1 - cur][i].x * sgn;
            }
            #pragma unroll
            for (int off = 16; off; off >>= 1) acc += __shfl_xor(acc, off, 32);
            if (lane == 0) qv[slot][k] = acc;
        }
        __syncthreads();
    }

    // ---- H_cat row: h'_e = s_e * relu(q_e @ rW_e + rb_e), zeros elsewhere
    for (int e = 0; e < Ee; ++e) {
        const float s = sv[e];
        const int slot = (e == sel[0]) ? 0 : ((e == sel[1]) ? 1 : -1);
        for (int h = tid; h < Hh; h += 256) {
            u16 outv = 0;
            if (slot >= 0) {
                float acc = rb[(size_t)e * Hh + h];
                #pragma unroll
                for (int j = 0; j < NQq; ++j)
                    acc += qv[slot][j] * rW[((size_t)e * NQq + j) * Hh + h];
                acc = fmaxf(acc, 0.f) * s;
                outv = f2bf(acc);
            }
            Hcat[(size_t)b * KCAT + e * Hh + h] = outv;
        }
    }
    if (tid < Ee) svals[(size_t)b * Ee + tid] = sv[tid];
}

// ---------------------------------------------------------------------------
// Kernel 2: C[M,N] = A[M,K] * B[N,K]^T, bf16 WMMA, double-buffered LDS with
// CDNA5 async loads (global_load_async_to_lds_b128 + ASYNCcnt pipeline).
// WG = 256 thr (8 wave32), 128x128 output tile, each wave 2x4 WMMA tiles.
// ---------------------------------------------------------------------------
#define TM 128
#define TN 128
#define TK 32

// Stage one 128x32 bf16 tile into LDS with per-lane async b128 copies.
// Exactly two UNCONDITIONAL chunks per thread (512 chunks / 256 threads) so
// the compiler emits no exec-mask guards around the async issues.
// => 2 async instructions per wave per call, 4 per (A,B) tile pair.
__device__ __forceinline__ void async_stage_tile(
    const u16* gbase, int rowStride, int rowBase, int k0, u16* lds, int tid)
{
    #pragma unroll
    for (int half = 0; half < 2; ++half) {
        const int i = tid + half * 256;          // 0..511, always in range
        const int r = i >> 2, c = (i & 3) * 8;
        unsigned lds_off = (unsigned)(size_t)&lds[r * TK + c];   // wave-relative LDS addr
        unsigned vofs = (unsigned)(((u32)(rowBase + r) * (u32)rowStride + (u32)(k0 + c))
                                   * (u32)sizeof(u16));
        asm volatile("global_load_async_to_lds_b128 %0, %1, %2"
                     :: "v"(lds_off), "v"(vofs), "s"((u64)gbase)
                     : "memory");
    }
}

__global__ __launch_bounds__(256) void gemm_bf16_wmma_kernel(
    const u16* __restrict__ A,    // [M][K] bf16 (H_cat)
    const u16* __restrict__ Bt,   // [N][K] bf16 (w2 N-major)
    float* __restrict__ C,        // [M][N] fp32
    int M, int N, int Kdim)
{
    __shared__ __align__(16) u16 As[2][TM * TK];
    __shared__ __align__(16) u16 Bs[2][TN * TK];

    const int tid   = threadIdx.x;
    const int lane  = tid & 31;
    const int wave  = tid >> 5;
    const int waveM = wave & 3;     // 4 waves along M -> 32 rows each
    const int waveN = wave >> 2;    // 2 waves along N -> 64 cols each
    const int mBase = blockIdx.y * TM;
    const int nBase = blockIdx.x * TN;

    v8f acc[2][4];
    #pragma unroll
    for (int mi = 0; mi < 2; ++mi)
        #pragma unroll
        for (int ni = 0; ni < 4; ++ni)
            #pragma unroll
            for (int j = 0; j < 8; ++j) acc[mi][ni][j] = 0.f;

    const int klo  = (lane < 16) ? 0 : 8;   // ISA 16-bit fragment layout
    const int mrow = lane & 15;

    union FragU { uint4 u[2]; v16bf v; };

    // prologue: tile 0 in flight (ASYNCcnt = 4 per wave)
    async_stage_tile(A,  Kdim, mBase, 0, As[0], tid);
    async_stage_tile(Bt, Kdim, nBase, 0, Bs[0], tid);

    int cur = 0;
    for (int k0 = 0; k0 < Kdim; k0 += TK) {
        const bool has_next = (k0 + TK) < Kdim;
        if (has_next) {                 // issue tile k+1 into other buffer
            async_stage_tile(A,  Kdim, mBase, k0 + TK, As[cur ^ 1], tid);
            async_stage_tile(Bt, Kdim, nBase, k0 + TK, Bs[cur ^ 1], tid);
            asm volatile("s_wait_asynccnt 4" ::: "memory");   // tile k landed
        } else {
            asm volatile("s_wait_asynccnt 0" ::: "memory");
        }
        __syncthreads();                // all waves' tile-k async copies visible

        const u16* Ac = As[cur];
        const u16* Bc = Bs[cur];
        v16bf afrag[2], bfrag[4];
        #pragma unroll
        for (int mi = 0; mi < 2; ++mi) {
            int row = waveM * 32 + mi * 16 + mrow;
            FragU U;
            U.u[0] = *(const uint4*)&Ac[row * TK + klo];
            U.u[1] = *(const uint4*)&Ac[row * TK + klo + 16];
            afrag[mi] = U.v;
        }
        #pragma unroll
        for (int ni = 0; ni < 4; ++ni) {
            int row = waveN * 64 + ni * 16 + mrow;
            FragU U;
            U.u[0] = *(const uint4*)&Bc[row * TK + klo];
            U.u[1] = *(const uint4*)&Bc[row * TK + klo + 16];
            bfrag[ni] = U.v;
        }

        #pragma unroll
        for (int mi = 0; mi < 2; ++mi)
            #pragma unroll
            for (int ni = 0; ni < 4; ++ni)
                acc[mi][ni] = __builtin_amdgcn_wmma_f32_16x16x32_bf16(
                    false, afrag[mi], false, bfrag[ni],
                    (short)0, acc[mi][ni], false, false);

        __syncthreads();                // everyone done reading buf[cur]
        cur ^= 1;
    }

    // write back per ISA f32 C/D layout: vgpr r, lanes>=16 -> M += 8
    #pragma unroll
    for (int mi = 0; mi < 2; ++mi) {
        #pragma unroll
        for (int ni = 0; ni < 4; ++ni) {
            int col  = nBase + waveN * 64 + ni * 16 + (lane & 15);
            int row0 = mBase + waveM * 32 + mi * 16 + ((lane >> 4) << 3);
            #pragma unroll
            for (int r = 0; r < 8; ++r)
                C[(size_t)(row0 + r) * N + col] = acc[mi][ni][r];
        }
    }
}

// ---------------------------------------------------------------------------
// Kernel 3: y = xin + gemm + sum_e s_e*b2_e ; LayerNorm(y)*g + beta
// ---------------------------------------------------------------------------
__global__ __launch_bounds__(256) void resid_ln_kernel(
    const float* __restrict__ xin,   // [B][D]
    const float* __restrict__ gemm,  // [B][D]
    const float* __restrict__ svals, // [B][E]
    const float* __restrict__ b2,    // [E][D]
    const float* __restrict__ g,     // [D]
    const float* __restrict__ beta,  // [D]
    float* __restrict__ out)         // [B][D]
{
    const int b    = blockIdx.x;
    const int tid  = threadIdx.x;
    const int lane = tid & 31;
    const int wave = tid >> 5;

    __shared__ float sv[Ee];
    __shared__ float ybuf[Dd];
    __shared__ float red[16];

    if (tid < Ee) sv[tid] = svals[(size_t)b * Ee + tid];
    __syncthreads();

    float sum = 0.f, sq = 0.f;
    for (int d = tid; d < Dd; d += 256) {
        float bias = 0.f;
        #pragma unroll
        for (int e = 0; e < Ee; ++e) bias += sv[e] * b2[(size_t)e * Dd + d];
        float y = xin[(size_t)b * Dd + d] + gemm[(size_t)b * Dd + d] + bias;
        ybuf[d] = y;
        sum += y; sq += y * y;
    }
    #pragma unroll
    for (int off = 16; off; off >>= 1) {
        sum += __shfl_xor(sum, off, 32);
        sq  += __shfl_xor(sq,  off, 32);
    }
    if (lane == 0) { red[wave] = sum; red[8 + wave] = sq; }
    __syncthreads();
    if (tid == 0) {
        float S = 0.f, Q = 0.f;
        for (int w = 0; w < 8; ++w) { S += red[w]; Q += red[8 + w]; }
        red[0] = S * (1.f / Dd);
        red[1] = Q * (1.f / Dd);
    }
    __syncthreads();
    const float mu   = red[0];
    const float var  = red[1] - mu * mu;
    const float rstd = rsqrtf(var + 1e-5f);
    for (int d = tid; d < Dd; d += 256)
        out[(size_t)b * Dd + d] = (ybuf[d] - mu) * rstd * g[d] + beta[d];
}

// ---------------------------------------------------------------------------
extern "C" void kernel_launch(void* const* d_in, const int* in_sizes, int n_in,
                              void* d_out, int out_size, void* d_ws, size_t ws_size,
                              hipStream_t stream) {
    const float* x   = (const float*)d_in[0];
    const float* gW  = (const float*)d_in[1];
    const float* gb  = (const float*)d_in[2];
    const float* qp  = (const float*)d_in[3];
    const float* rW  = (const float*)d_in[4];
    const float* rb  = (const float*)d_in[5];
    const float* w2  = (const float*)d_in[6];
    const float* b2  = (const float*)d_in[7];
    const float* lng = (const float*)d_in[8];
    const float* lnb = (const float*)d_in[9];
    float* out = (float*)d_out;

    char* ws = (char*)d_ws;
    size_t off = 0;
    u16*   Hcat  = (u16*)(ws + off);   off += (size_t)Bsz * KCAT * sizeof(u16);      // 128 MB
    u16*   w2bf  = (u16*)(ws + off);   off += (size_t)NBb * Dd * KCAT * sizeof(u16); //  32 MB
    float* gout  = (float*)(ws + off); off += (size_t)Bsz * Dd * sizeof(float);      //  32 MB
    float* svals = (float*)(ws + off); off += (size_t)Bsz * Ee * sizeof(float);
    float* ybuf  = (float*)(ws + off); off += (size_t)Bsz * Dd * sizeof(float);      //  32 MB

    // one-time weight conversion/transpose (coalesced both directions)
    {
        dim3 cg(Dd / 64, Hh / 64, NBb * Ee);   // (16,16,16)
        conv_w2_kernel<<<cg, 256, 0, stream>>>(w2, w2bf);
    }

    const float* cur = x;
    for (int blk = 0; blk < NBb; ++blk) {
        gate_qsim_h_kernel<<<Bsz, 256, 0, stream>>>(
            cur,
            gW + (size_t)blk * Dd * Ee,
            gb + (size_t)blk * Ee,
            qp + (size_t)blk * Ee * NLl * NQq * 3,
            rW + (size_t)blk * Ee * NQq * Hh,
            rb + (size_t)blk * Ee * Hh,
            Hcat, svals);

        dim3 gg(Dd / TN, Bsz / TM);   // (8, 64)
        gemm_bf16_wmma_kernel<<<gg, 256, 0, stream>>>(
            Hcat, w2bf + (size_t)blk * Dd * KCAT, gout, Bsz, Dd, KCAT);

        float* dst = (blk == NBb - 1) ? out : ybuf;
        resid_ln_kernel<<<Bsz, 256, 0, stream>>>(
            cur, gout, svals,
            b2 + (size_t)blk * Ee * Dd,
            lng + (size_t)blk * Dd,
            lnb + (size_t)blk * Dd,
            dst);
        cur = dst;
    }
}
